// TransformerEncoder_6296422056642
// MI455X (gfx1250) — matrix operand order repaired and checked by reading, not compiled, not run on hardware
//
#include <hip/hip_runtime.h>
#include <hip/hip_bf16.h>

#define EMBED 768
#define HEADS 12
#define HDIM  64
#define SEQ   2048
#define BATCH 2
#define NTOK  (BATCH * SEQ)
#define FFDIM 3072

typedef __attribute__((ext_vector_type(8)))  __bf16 v8bf;
typedef __attribute__((ext_vector_type(16))) __bf16 v16bf;
typedef __attribute__((ext_vector_type(8)))  float  v8f;
typedef __attribute__((ext_vector_type(4)))  int    v4i;

typedef __attribute__((address_space(1))) v4i g_v4i;
typedef __attribute__((address_space(3))) v4i l_v4i;

enum { MODE_HEAD_QK = 0, MODE_HEAD_VT = 1, MODE_RESID_F32 = 2, MODE_RELU_BF16 = 3 };

#define HAS_ASYNC_LDS __has_builtin(__builtin_amdgcn_global_load_async_to_lds_b128)

// ---------------------------------------------------------------------------
// Async global -> LDS 128-bit copy (per-lane), tracked by ASYNCcnt.
// Builtin prototype (from hipcc diagnostic): (v4i AS1*, v4i AS3*, imm, imm).
// ---------------------------------------------------------------------------
__device__ inline void copy_b128_to_lds(const __bf16* g, __bf16* l) {
#if HAS_ASYNC_LDS
  __builtin_amdgcn_global_load_async_to_lds_b128(
      (g_v4i*)(uintptr_t)g, (l_v4i*)(uint32_t)(uintptr_t)l, 0, 0);
#else
  *(v8bf*)l = *(const v8bf*)g;
#endif
}

template <int N> __device__ inline void wait_asynccnt() {
#if HAS_ASYNC_LDS
#if __has_builtin(__builtin_amdgcn_s_wait_asynccnt)
  __builtin_amdgcn_s_wait_asynccnt(N);
#else
  asm volatile("s_wait_asynccnt %0" ::"i"(N));
#endif
#endif
}

// ---------------------------------------------------------------------------
// WMMA fragment loaders (bf16, 16x16x32).
// A (16x32, MxK): lane L holds row M=L%16; K = {k + (L/16)*8 + 0..7} and
//                 {k+16 + (L/16)*8 + 0..7}  (ISA 7.12.2 16-bit A layout).
// B (32x16, KxN): lane L holds column N=L%16; K = {k + (L/16)*8 + 0..7} and
//                 {k+16 + (L/16)*8 + 0..7}  (same K order as A).
// Both are two 128-bit contiguous loads per lane.
// ---------------------------------------------------------------------------
__device__ inline v16bf load_a_frag(const __bf16* __restrict__ base, int ld,
                                    int lr, int half, int k) {
  const __bf16* p = base + (size_t)lr * ld + k + half * 8;
  union { v16bf v; v8bf h[2]; } u;
  u.h[0] = *(const v8bf*)p;
  u.h[1] = *(const v8bf*)(p + 16);
  return u.v;
}

__device__ inline v16bf load_b_frag(const __bf16* __restrict__ base, int ld,
                                    int lr, int half, int k) {
  const __bf16* p = base + (size_t)lr * ld + k + half * 8;
  union { v16bf v; v8bf h[2]; } u;
  u.h[0] = *(const v8bf*)p;
  u.h[1] = *(const v8bf*)(p + 16);
  return u.v;
}

__device__ inline v8f wmma_bf16(v16bf a, v16bf b, v8f c) {
  return __builtin_amdgcn_wmma_f32_16x16x32_bf16(false, a, false, b,
                                                 (short)0, c, false, false);
}

__device__ inline float groupmax16(float v) {
  v = fmaxf(v, __shfl_xor(v, 1, 32));
  v = fmaxf(v, __shfl_xor(v, 2, 32));
  v = fmaxf(v, __shfl_xor(v, 4, 32));
  v = fmaxf(v, __shfl_xor(v, 8, 32));
  return v;
}

// ---------------------------------------------------------------------------
// Weight transpose + f32->bf16 convert: W (KxN row-major) -> WT (NxK row-major)
// ---------------------------------------------------------------------------
__global__ __launch_bounds__(256) void transpose_f32_to_bf16(
    const float* __restrict__ W, __bf16* __restrict__ WT, int K, int N) {
  __shared__ float tile[32][33];
  int n0 = blockIdx.x * 32, k0 = blockIdx.y * 32;
  int tx = threadIdx.x, ty = threadIdx.y;  // (32, 8)
  #pragma unroll
  for (int i = ty; i < 32; i += 8)
    tile[i][tx] = W[(size_t)(k0 + i) * N + n0 + tx];
  __syncthreads();
  #pragma unroll
  for (int i = ty; i < 32; i += 8)
    WT[(size_t)(n0 + i) * K + k0 + tx] = (__bf16)tile[tx][i];
}

// ---------------------------------------------------------------------------
// LayerNorm over EMBED=768; one block (256 threads, 3 elems/thread) per token.
// ---------------------------------------------------------------------------
__global__ __launch_bounds__(256) void ln_kernel(
    const float* __restrict__ x, const float* __restrict__ g,
    const float* __restrict__ b, __bf16* __restrict__ out_bf,
    float* __restrict__ out_f) {
  __shared__ float red[256];
  const int t = threadIdx.x;
  const size_t base = (size_t)blockIdx.x * EMBED;
  float v0 = x[base + t], v1 = x[base + t + 256], v2 = x[base + t + 512];
  red[t] = v0 + v1 + v2;
  __syncthreads();
  for (int o = 128; o > 0; o >>= 1) {
    if (t < o) red[t] += red[t + o];
    __syncthreads();
  }
  float mu = red[0] * (1.0f / EMBED);
  __syncthreads();
  float d0 = v0 - mu, d1 = v1 - mu, d2 = v2 - mu;
  red[t] = d0 * d0 + d1 * d1 + d2 * d2;
  __syncthreads();
  for (int o = 128; o > 0; o >>= 1) {
    if (t < o) red[t] += red[t + o];
    __syncthreads();
  }
  float r = rsqrtf(red[0] * (1.0f / EMBED) + 1e-5f);
  float y0 = d0 * r * g[t] + b[t];
  float y1 = d1 * r * g[t + 256] + b[t + 256];
  float y2 = d2 * r * g[t + 512] + b[t + 512];
  if (out_bf) {
    out_bf[base + t] = (__bf16)y0;
    out_bf[base + t + 256] = (__bf16)y1;
    out_bf[base + t + 512] = (__bf16)y2;
  }
  if (out_f) {
    out_f[base + t] = y0;
    out_f[base + t + 256] = y1;
    out_f[base + t + 512] = y2;
  }
}

// ---------------------------------------------------------------------------
// Dense GEMM: C(MxN) = A(MxK, bf16 row-major) x WT(NxK, bf16 row-major)^T.
// Block = 4 waves = 128x64 output tile. The 64x32 B tile is staged once per
// block into LDS with GLOBAL_LOAD_ASYNC_TO_LDS_B128 (double-buffered,
// ASYNCcnt-tracked) and shared by all 4 waves; A streams from global (L2).
// Each wave: 2x4 WMMA C-frags -> 8 v_wmma_f32_16x16x32_bf16 per K-step.
// B rows padded to 40 bf16 (80 B) -> 16 lane reads hit distinct bank quads.
// ---------------------------------------------------------------------------
template <int MODE>
__global__ __launch_bounds__(128) void gemm_wmma(
    const __bf16* __restrict__ A, const __bf16* __restrict__ WT,
    int M, int N, int K, const float* __restrict__ bias,
    const float* __restrict__ resid, void* __restrict__ outp) {
  __shared__ __align__(16) __bf16 Bs[2][64][40];
  const int tid  = threadIdx.x;
  const int lane = tid & 31;
  const int wid  = tid >> 5;
  const int half = lane >> 4;
  const int lr   = lane & 15;
  const int m0 = (blockIdx.x * 4 + wid) * 32;
  const int n0 = blockIdx.y * 64;

  v8f c[2][4];
  const v8f z = {0.f, 0.f, 0.f, 0.f, 0.f, 0.f, 0.f, 0.f};
  #pragma unroll
  for (int rt = 0; rt < 2; rt++)
    #pragma unroll
    for (int ct = 0; ct < 4; ct++) c[rt][ct] = z;

  const __bf16* Abase = A + (size_t)m0 * K;
  const __bf16* Bbase = WT + (size_t)n0 * K;

  // Stage one 64x32 B tile (4 KB) into Bs[buf]: 256 x b128, 2 per thread.
  auto stage = [&](int buf, int k) {
    #pragma unroll
    for (int i = 0; i < 2; i++) {
      const int t = tid + i * 128;       // 0..255
      const int row = t >> 2, seg = t & 3;
      copy_b128_to_lds(Bbase + (size_t)row * K + k + seg * 8,
                       &Bs[buf][row][seg * 8]);
    }
  };

  const int NK = K >> 5;
  stage(0, 0);
  for (int kt = 0; kt < NK; ++kt) {
    const int cur = kt & 1;
    const int k = kt << 5;
    if (kt + 1 < NK) {
      stage(cur ^ 1, k + 32);
      wait_asynccnt<2>();   // my 2 oldest (current tile) done
    } else {
      wait_asynccnt<0>();
    }
    __syncthreads();        // everyone's current-tile copies visible

    v16bf a[2], bfr[4];
    a[0] = load_a_frag(Abase, K, lr, half, k);
    a[1] = load_a_frag(Abase + (size_t)16 * K, K, lr, half, k);
    #pragma unroll
    for (int ct = 0; ct < 4; ct++)
      bfr[ct] = load_b_frag(&Bs[cur][ct * 16][0], 40, lr, half, 0);
    #pragma unroll
    for (int rt = 0; rt < 2; rt++)
      #pragma unroll
      for (int ct = 0; ct < 4; ct++)
        c[rt][ct] = wmma_bf16(a[rt], bfr[ct], c[rt][ct]);

    __syncthreads();        // all waves done reading Bs[cur] before overwrite
  }

  #pragma unroll
  for (int rt = 0; rt < 2; rt++) {
    #pragma unroll
    for (int ct = 0; ct < 4; ct++) {
      #pragma unroll
      for (int j = 0; j < 8; j++) {
        const int m = m0 + rt * 16 + j + half * 8;
        const int n = n0 + ct * 16 + lr;
        float v = c[rt][ct][j] + bias[n];
        if (MODE == MODE_RESID_F32) {
          float* out = (float*)outp;
          const size_t idx = (size_t)m * N + n;
          out[idx] = resid[idx] + v;
        } else if (MODE == MODE_RELU_BF16) {
          __bf16* out = (__bf16*)outp;
          out[(size_t)m * N + n] = (__bf16)fmaxf(v, 0.f);
        } else if (MODE == MODE_HEAD_QK) {
          __bf16* out = (__bf16*)outp;
          const int bb = m / SEQ, s = m % SEQ, h = n / HDIM, d = n % HDIM;
          out[(((size_t)(bb * HEADS + h)) * SEQ + s) * HDIM + d] = (__bf16)v;
        } else {  // MODE_HEAD_VT: V stored transposed [BH][D][S]
          __bf16* out = (__bf16*)outp;
          const int bb = m / SEQ, s = m % SEQ, h = n / HDIM, d = n % HDIM;
          out[(((size_t)(bb * HEADS + h)) * HDIM + d) * SEQ + s] = (__bf16)v;
        }
      }
    }
  }
}

// ---------------------------------------------------------------------------
// Flash attention, per-wave 16-row Q tile, KV chunks of 64, online softmax.
// scores = Q(16x64) x K^T via 4x2 wmma. Row max: cross-lane shuffle reduce
// (32 ds_bpermute / chunk). Row sum: NO shuffles -- l += P x ones via WMMA,
// which lands the row sum broadcast across the C-frag columns in exactly the
// accumulator layout. ctx += P(16x64) x V(64x64) via 2x4 wmma on
// transposed-V B-frags. P relayout through a 2.25KB/wave LDS bounce
// (rows padded to 72 bf16 -> conflict-free 16B lane reads).
// ---------------------------------------------------------------------------
__global__ __launch_bounds__(128) void attn_kernel(
    const __bf16* __restrict__ Q, const __bf16* __restrict__ Kb,
    const __bf16* __restrict__ VT, __bf16* __restrict__ ctx) {
  __shared__ __align__(16) __bf16 sp[4][16][72];
  const int lane = threadIdx.x & 31;
  const int wid  = threadIdx.x >> 5;
  const int half = lane >> 4;
  const int lr   = lane & 15;
  const int bh = blockIdx.y;
  const int s0 = (blockIdx.x * 4 + wid) * 16;
  const float scale = rsqrtf((float)EMBED);  // reference scales by sqrt(embed)

  const __bf16* q  = Q + ((size_t)bh * SEQ + s0) * HDIM;
  const __bf16* kk = Kb + (size_t)bh * SEQ * HDIM;
  const __bf16* vt = VT + (size_t)bh * HDIM * SEQ;

  const v16bf aq0 = load_a_frag(q, HDIM, lr, half, 0);
  const v16bf aq1 = load_a_frag(q, HDIM, lr, half, 32);

  v16bf ones;
  #pragma unroll
  for (int i = 0; i < 16; i++) ones[i] = (__bf16)1.0f;

  float mrow[8];
  v8f acc[4], lacc;
  const v8f z = {0.f, 0.f, 0.f, 0.f, 0.f, 0.f, 0.f, 0.f};
  #pragma unroll
  for (int j = 0; j < 8; j++) mrow[j] = -1e30f;
  #pragma unroll
  for (int ct = 0; ct < 4; ct++) acc[ct] = z;
  lacc = z;

  for (int t0 = 0; t0 < SEQ; t0 += 64) {
    // ---- scores: 16 x 64 tile, 8 wmma ----
    v8f sc[4];
    #pragma unroll
    for (int ct = 0; ct < 4; ct++) {
      const __bf16* kb = kk + (size_t)(t0 + ct * 16) * HDIM;
      v16bf b0 = load_b_frag(kb, HDIM, lr, half, 0);
      sc[ct] = wmma_bf16(aq0, b0, z);
      v16bf b1 = load_b_frag(kb, HDIM, lr, half, 32);
      sc[ct] = wmma_bf16(aq1, b1, sc[ct]);
    }
    #pragma unroll
    for (int ct = 0; ct < 4; ct++)
      #pragma unroll
      for (int j = 0; j < 8; j++) sc[ct][j] *= scale;

    // ---- online softmax: row max via shuffle reduce, P -> LDS ----
    float f[8];
    #pragma unroll
    for (int j = 0; j < 8; j++) {
      float mx = fmaxf(fmaxf(sc[0][j], sc[1][j]), fmaxf(sc[2][j], sc[3][j]));
      mx = groupmax16(mx);
      const float mn = fmaxf(mrow[j], mx);
      f[j] = __expf(mrow[j] - mn);
      mrow[j] = mn;
      const int row = j + half * 8;
      #pragma unroll
      for (int ct = 0; ct < 4; ct++) {
        const float p = __expf(sc[ct][j] - mn);
        sp[wid][row][ct * 16 + lr] = (__bf16)p;
      }
    }
    // rescale running accumulators (ctx and row-sum) by exp(m_old - m_new)
    #pragma unroll
    for (int ct = 0; ct < 4; ct++)
      #pragma unroll
      for (int j = 0; j < 8; j++) acc[ct][j] *= f[j];
    #pragma unroll
    for (int j = 0; j < 8; j++) lacc[j] *= f[j];

    // ---- reload P in A-frag layout ----
    const v16bf pa0 = load_a_frag(&sp[wid][0][0], 72, lr, half, 0);
    const v16bf pa1 = load_a_frag(&sp[wid][0][0], 72, lr, half, 32);

    // ---- row sums via WMMA with all-ones B (no cross-lane shuffles) ----
    lacc = wmma_bf16(pa0, ones, lacc);
    lacc = wmma_bf16(pa1, ones, lacc);

    // ---- ctx += P x V: 8 wmma ----
    #pragma unroll
    for (int ct = 0; ct < 4; ct++) {
      const __bf16* vb = vt + (size_t)(ct * 16) * SEQ + t0;
      v16bf bv0 = load_b_frag(vb, SEQ, lr, half, 0);
      acc[ct] = wmma_bf16(pa0, bv0, acc[ct]);
      v16bf bv1 = load_b_frag(vb, SEQ, lr, half, 32);
      acc[ct] = wmma_bf16(pa1, bv1, acc[ct]);
    }
  }

  const int bb = bh / HEADS, hh = bh % HEADS;
  #pragma unroll
  for (int ct = 0; ct < 4; ct++)
    #pragma unroll
    for (int j = 0; j < 8; j++) {
      const int row = j + half * 8;
      const float o = acc[ct][j] / lacc[j];
      ctx[((size_t)(bb * SEQ) + s0 + row) * EMBED + hh * HDIM + ct * 16 + lr] =
          (__bf16)o;
    }
}

// ---------------------------------------------------------------------------
extern "C" void kernel_launch(void* const* d_in, const int* in_sizes, int n_in,
                              void* d_out, int out_size, void* d_ws,
                              size_t ws_size, hipStream_t stream) {
  const float* x     = (const float*)d_in[0];
  const float* Wq    = (const float*)d_in[1];
  const float* bq    = (const float*)d_in[2];
  const float* Wk    = (const float*)d_in[3];
  const float* bk    = (const float*)d_in[4];
  const float* Wv    = (const float*)d_in[5];
  const float* bv    = (const float*)d_in[6];
  const float* Wo    = (const float*)d_in[7];
  const float* bo    = (const float*)d_in[8];
  const float* ln1_g = (const float*)d_in[9];
  const float* ln1_b = (const float*)d_in[10];
  const float* ln2_g = (const float*)d_in[11];
  const float* ln2_b = (const float*)d_in[12];
  const float* W1    = (const float*)d_in[13];
  const float* b1    = (const float*)d_in[14];
  const float* W2    = (const float*)d_in[15];
  const float* b2    = (const float*)d_in[16];

  char* ws = (char*)d_ws;
  size_t off = 0;
  auto alloc = [&](size_t bytes) {
    void* p = ws + off;
    off = (off + bytes + 255) & ~(size_t)255;
    return p;
  };
  __bf16* WqT  = (__bf16*)alloc((size_t)EMBED * EMBED * 2);
  __bf16* WkT  = (__bf16*)alloc((size_t)EMBED * EMBED * 2);
  __bf16* WvT  = (__bf16*)alloc((size_t)EMBED * EMBED * 2);
  __bf16* WoT  = (__bf16*)alloc((size_t)EMBED * EMBED * 2);
  __bf16* W1T  = (__bf16*)alloc((size_t)EMBED * FFDIM * 2);
  __bf16* W2T  = (__bf16*)alloc((size_t)EMBED * FFDIM * 2);
  __bf16* hbf  = (__bf16*)alloc((size_t)NTOK * EMBED * 2);
  __bf16* qb   = (__bf16*)alloc((size_t)NTOK * EMBED * 2);
  __bf16* kbuf = (__bf16*)alloc((size_t)NTOK * EMBED * 2);
  __bf16* vtb  = (__bf16*)alloc((size_t)NTOK * EMBED * 2);
  __bf16* ctxb = (__bf16*)alloc((size_t)NTOK * EMBED * 2);
  float*  out1 = (float*)alloc((size_t)NTOK * EMBED * 4);
  __bf16* h2bf = (__bf16*)alloc((size_t)NTOK * EMBED * 2);
  __bf16* ff1  = (__bf16*)alloc((size_t)NTOK * FFDIM * 2);
  (void)ws_size; (void)in_sizes; (void)n_in; (void)out_size;

  const dim3 tb(32, 8);
  transpose_f32_to_bf16<<<dim3(EMBED / 32, EMBED / 32), tb, 0, stream>>>(Wq, WqT, EMBED, EMBED);
  transpose_f32_to_bf16<<<dim3(EMBED / 32, EMBED / 32), tb, 0, stream>>>(Wk, WkT, EMBED, EMBED);
  transpose_f32_to_bf16<<<dim3(EMBED / 32, EMBED / 32), tb, 0, stream>>>(Wv, WvT, EMBED, EMBED);
  transpose_f32_to_bf16<<<dim3(EMBED / 32, EMBED / 32), tb, 0, stream>>>(Wo, WoT, EMBED, EMBED);
  transpose_f32_to_bf16<<<dim3(FFDIM / 32, EMBED / 32), tb, 0, stream>>>(W1, W1T, EMBED, FFDIM);
  transpose_f32_to_bf16<<<dim3(EMBED / 32, FFDIM / 32), tb, 0, stream>>>(W2, W2T, FFDIM, EMBED);

  ln_kernel<<<NTOK, 256, 0, stream>>>(x, ln1_g, ln1_b, hbf, nullptr);

  const dim3 gE(NTOK / 128, EMBED / 64);
  gemm_wmma<MODE_HEAD_QK><<<gE, 128, 0, stream>>>(hbf, WqT, NTOK, EMBED, EMBED, bq, nullptr, qb);
  gemm_wmma<MODE_HEAD_QK><<<gE, 128, 0, stream>>>(hbf, WkT, NTOK, EMBED, EMBED, bk, nullptr, kbuf);
  gemm_wmma<MODE_HEAD_VT><<<gE, 128, 0, stream>>>(hbf, WvT, NTOK, EMBED, EMBED, bv, nullptr, vtb);

  attn_kernel<<<dim3(SEQ / 64, BATCH * HEADS), 128, 0, stream>>>(qb, kbuf, vtb, ctxb);

  gemm_wmma<MODE_RESID_F32><<<gE, 128, 0, stream>>>(ctxb, WoT, NTOK, EMBED, EMBED, bo, x, out1);

  ln_kernel<<<NTOK, 256, 0, stream>>>(out1, ln2_g, ln2_b, h2bf, (float*)d_out);

  gemm_wmma<MODE_RELU_BF16><<<dim3(NTOK / 128, FFDIM / 64), 128, 0, stream>>>(
      h2bf, W1T, NTOK, FFDIM, EMBED, b1, nullptr, ff1);
  gemm_wmma<MODE_RESID_F32><<<gE, 128, 0, stream>>>(
      ff1, W2T, NTOK, EMBED, FFDIM, b2, (const float*)d_out, d_out);
}
